// RNNAgentDIVA_61040075211058
// MI455X (gfx1250) — matrix operand order repaired and checked by reading, not compiled, not run on hardware
//
#include <hip/hip_runtime.h>
#include <hip/hip_bf16.h>
#include <math.h>

// Problem sizes (compile-time constants from the reference)
#define B_ROWS 32768
#define IN_DIM 512
#define H_DIM  1024
#define A_DIM  128
#define RMAXV  10.0f

typedef unsigned short u16;
typedef __attribute__((ext_vector_type(16))) __bf16 v16bf;
typedef __attribute__((ext_vector_type(8)))  float  v8f;
typedef __attribute__((ext_vector_type(4)))  unsigned v4u;
typedef __attribute__((ext_vector_type(8)))  int      v8i;
typedef __attribute__((ext_vector_type(4)))  int      v4i;

union FragBF { uint4 u[2]; v16bf v; };

#define LDSPITCH 40   // u16 elems per LDS tile row (80B): 64B data + 16B TDM pad

__device__ __forceinline__ u16 f32_to_bf16_rne(float x) {
  unsigned u = __float_as_uint(x);
  u += 0x7FFFu + ((u >> 16) & 1u);   // round-to-nearest-even
  return (u16)(u >> 16);
}

// ---- A-fragment (16x32 bf16, M x K), CDNA5 layout:
// lane l (0..15): row M=l; holds K = {8h..8h+7, 16+8h..16+8h+7}, h = lane>>4.
// `p` = per-lane row pointer already offset by (+row*ld + 8h); k-chunk added by caller.
__device__ __forceinline__ v16bf frag_a_from(const u16* p) {
  FragBF f;
  f.u[0] = *(const uint4*)(p);        // K = 8h .. 8h+7
  f.u[1] = *(const uint4*)(p + 16);   // K = 8h+16 .. +7
  return f.v;
}

// ---- B-fragment (32x16 bf16, K x N): lane n = lane&15 is the output column,
// holds K = {8h..8h+7, 16+8h..16+8h+7}. `p` = per-lane pointer at (col row base + kb).
__device__ __forceinline__ v16bf frag_b_from(const u16* p) {
  FragBF f;
  f.u[0] = *(const uint4*)(p);
  f.u[1] = *(const uint4*)(p + 16);
  return f.v;
}

__device__ __forceinline__ v8f wmma_bf16(v16bf a, v16bf b, v8f c) {
  // (neg_a, A, neg_b, B, c_mod, C, reuse_a, reuse_b)
  return __builtin_amdgcn_wmma_f32_16x16x32_bf16(false, a, false, b, (short)0, c,
                                                 false, false);
}

// ---------------------------------------------------------------------------
// Tensor Data Mover: DMA one 2D tile (rows x 32 bf16 elems) global -> LDS.
// D# per CDNA5 ISA 8.3-8.6. Padding fields insert 16B after every 64B so the
// LDS tile lands with an 80B row pitch (bank-conflict-free fragment reads).
// Issued by a single wave; tracked with TENSORcnt (in-order per wave).
// NOTE: this toolchain (clang-23 lane) exposes the 6-arg builtin:
//   (uint32x4 g0, int32x8 g1, int32x4 g2, int32x4 g3, int32x8 g4, i32 cpol)
__device__ __forceinline__ void tdm_load_tile_2d(const u16* gsrc, unsigned lds_off,
                                                 int rows, int pitch_elems) {
  const unsigned long long ga = (unsigned long long)(size_t)gsrc;
  const unsigned TDBIG = 0x40000000u;   // huge tensor dims: tiles never go OOB
  v4u g0;
  g0.x = 1u;                                        // count=1, user mode, no gather
  g0.y = lds_off;                                   // D#.lds_addr (bytes)
  g0.z = (unsigned)ga;                              // D#.global_addr[31:0]
  g0.w = (unsigned)((ga >> 32) & 0x01FFFFFFu)       // D#.global_addr[56:32]
       | (2u << 30);                                // D#.type = 2 ("image")
  v8i g1;
  g1[0] = (1 << 16)                                 // data_size = 1 -> 2 bytes
        | (1 << 20)                                 // pad_enable
        | (3 << 22)                                 // pad_interval = 3 -> every 64B
        | (3 << 25);                                // pad_amount = 3 -> 4 DW = 16B
  g1[1] = (int)((TDBIG & 0xFFFFu) << 16);           // tensor_dim0[15:0] (hi half)
  g1[2] = (int)(((TDBIG >> 16) & 0xFFFFu)           // tensor_dim0[31:16]
        | ((TDBIG & 0xFFFFu) << 16));               // tensor_dim1[15:0]
  g1[3] = (int)(((TDBIG >> 16) & 0xFFFFu)           // tensor_dim1[31:16]
        | (32u << 16));                             // tile_dim0 = 32 (k-chunk)
  g1[4] = rows;                                     // tile_dim1 = rows; tile_dim2 = 0
  g1[5] = pitch_elems;                              // tensor_dim0_stride[31:0]
  g1[6] = 0;                                        // stride[47:32]=0; dim1_stride lo
  g1[7] = 0;                                        // dim1_stride hi
  v4i g2 = {0, 0, 0, 0};                            // 2D tile: groups 2/3 unused
  v4i g3 = {0, 0, 0, 0};
  v8i g4 = {0, 0, 0, 0, 0, 0, 0, 0};                // extra group (zero-filled)
  __builtin_amdgcn_tensor_load_to_lds(g0, g1, g2, g3, g4, /*cpol=*/0);
}

__device__ __forceinline__ float sigmoidf_fast(float x) {
  return 1.0f / (1.0f + __expf(-x));
}

// ---------------------------------------------------------------------------
// fp32 -> bf16 conversion (one-shot, bandwidth-bound; n must be multiple of 4)
__global__ void cvt_f32_to_bf16_kernel(const float* __restrict__ src,
                                       u16* __restrict__ dst, int n4) {
  int i = blockIdx.x * blockDim.x + threadIdx.x;
  if (i < n4) {
    float4 f = ((const float4*)src)[i];
    unsigned lo = (unsigned)f32_to_bf16_rne(f.x) | ((unsigned)f32_to_bf16_rne(f.y) << 16);
    unsigned hi = (unsigned)f32_to_bf16_rne(f.z) | ((unsigned)f32_to_bf16_rne(f.w) << 16);
    ((uint2*)dst)[i] = make_uint2(lo, hi);
  }
}

// ---------------------------------------------------------------------------
// Stage 1: x = relu(inputs @ W1.T + b1), write bf16 x to workspace.
// Block = 256 thr = 8 waves; block tile 128(M) x 128(N); wave tile 16 x 128.
// W1 tile (128 x 32 bf16 per chunk) staged by the TDM, double buffered.
__global__ __launch_bounds__(256) void fc1_relu_kernel(
    const u16* __restrict__ in_bf, const u16* __restrict__ W1bf,
    const float* __restrict__ b1, u16* __restrict__ x_bf) {
  __shared__ __align__(128) u16 sW[2][128][LDSPITCH];   // 20,480 B

  const int wave = threadIdx.x >> 5;
  const int lane = threadIdx.x & 31;
  const int l = lane & 15, h = lane >> 4;
  const int m0 = blockIdx.x * 128 + wave * 16;
  const int n0 = blockIdx.y * 128;

  const u16* pa = in_bf + (size_t)(m0 + l) * IN_DIM + 8 * h;  // A row base
  const int kb = h << 3;                                       // B half select

  if (wave == 0)
    tdm_load_tile_2d(W1bf + (size_t)n0 * IN_DIM, (unsigned)(size_t)&sW[0][0][0],
                     128, IN_DIM);

  v8f acc[8] = {};
  int buf = 0;
  for (int kc = 0; kc < IN_DIM; kc += 32) {
    const bool has_next = (kc + 32) < IN_DIM;
    if (wave == 0) {
      if (has_next) {
        tdm_load_tile_2d(W1bf + (size_t)n0 * IN_DIM + kc + 32,
                         (unsigned)(size_t)&sW[buf ^ 1][0][0], 128, IN_DIM);
        __builtin_amdgcn_s_wait_tensorcnt(1);   // in-order: current buf is done
      } else {
        __builtin_amdgcn_s_wait_tensorcnt(0);
      }
    }
    __syncthreads();                            // buf visible to all waves

    v16bf a = frag_a_from(pa + kc);
    v16bf bf[8];
#pragma unroll
    for (int s = 0; s < 8; ++s) bf[s] = frag_b_from(&sW[buf][16 * s + l][kb]);
#pragma unroll
    for (int s = 0; s < 8; ++s) acc[s] = wmma_bf16(a, bf[s], acc[s]);

    __syncthreads();                            // done reading buf before restage
    buf ^= 1;
  }

#pragma unroll
  for (int s = 0; s < 8; ++s) {
    const int n = n0 + 16 * s + l;
    const float bias = b1[n];
#pragma unroll
    for (int j = 0; j < 8; ++j) {
      const int m = m0 + j + 8 * h;            // C/D layout: VGPR j -> row j(+8h)
      float v = acc[s][j] + bias;
      v = v > 0.0f ? v : 0.0f;
      x_bf[(size_t)m * H_DIM + n] = f32_to_bf16_rne(v);
    }
  }
}

// ---------------------------------------------------------------------------
// Stage 2: fused GRU cell.
// Block = 256 thr = 8 waves stacked in M: block tile 128(M) x 64(N).
// Six weight tiles per k-chunk (6 x 64 x 32 bf16) staged by six TDM
// descriptors, double buffered so the DMA of chunk k+1 overlaps the 24 WMMAs
// of chunk k.
__global__ __launch_bounds__(256) void gru_fused_kernel(
    const u16* __restrict__ x_bf, const u16* __restrict__ hid_bf,
    const float* __restrict__ hidden,
    const u16* __restrict__ Wxr, const u16* __restrict__ Whr,
    const u16* __restrict__ Wxz, const u16* __restrict__ Whz,
    const u16* __restrict__ Wxn, const u16* __restrict__ Whn,
    const float* __restrict__ bxr, const float* __restrict__ bhr,
    const float* __restrict__ bxz, const float* __restrict__ bhz,
    const float* __restrict__ bxn, const float* __restrict__ bhn,
    float* __restrict__ h_out, u16* __restrict__ h_bf) {
  __shared__ __align__(128) u16 smem[2][6][64][LDSPITCH];   // 61,440 B of 320KB

  const int wave = threadIdx.x >> 5;
  const int lane = threadIdx.x & 31;
  const int l = lane & 15, h = lane >> 4;
  const int m0 = blockIdx.x * 128 + wave * 16;
  const int n0 = blockIdx.y * 64;

  const u16* Wm[6] = {Wxr, Whr, Wxz, Whz, Wxn, Whn};

  // Per-lane A row bases and B half select.
  const u16* pax = x_bf   + (size_t)(m0 + l) * H_DIM + 8 * h;
  const u16* pah = hid_bf + (size_t)(m0 + l) * H_DIM + 8 * h;
  const int kb = h << 3;

  v8f accr[4] = {}, accz[4] = {}, accnx[4] = {}, acchn[4] = {};

  if (wave == 0) {
#pragma unroll
    for (int i = 0; i < 6; ++i)
      tdm_load_tile_2d(Wm[i] + (size_t)n0 * H_DIM,
                       (unsigned)(size_t)&smem[0][i][0][0], 64, H_DIM);
  }

  int buf = 0;
  for (int kc = 0; kc < H_DIM; kc += 32) {
    const bool has_next = (kc + 32) < H_DIM;
    if (wave == 0) {
      if (has_next) {
#pragma unroll
        for (int i = 0; i < 6; ++i)
          tdm_load_tile_2d(Wm[i] + (size_t)n0 * H_DIM + kc + 32,
                           (unsigned)(size_t)&smem[buf ^ 1][i][0][0], 64, H_DIM);
        __builtin_amdgcn_s_wait_tensorcnt(6);   // in-order: current 6 are done
      } else {
        __builtin_amdgcn_s_wait_tensorcnt(0);
      }
    }
    __syncthreads();                            // staged buf visible to all waves

    v16bf ax = frag_a_from(pax + kc);
    v16bf ah = frag_a_from(pah + kc);
#pragma unroll
    for (int s = 0; s < 4; ++s) {
      const int nr = 16 * s + l;   // LDS tile row (output column)
      v16bf br0 = frag_b_from(&smem[buf][0][nr][kb]);
      v16bf br1 = frag_b_from(&smem[buf][1][nr][kb]);
      v16bf bz0 = frag_b_from(&smem[buf][2][nr][kb]);
      v16bf bz1 = frag_b_from(&smem[buf][3][nr][kb]);
      v16bf bn0 = frag_b_from(&smem[buf][4][nr][kb]);
      v16bf bn1 = frag_b_from(&smem[buf][5][nr][kb]);
      accr[s]  = wmma_bf16(ax, br0, accr[s]);
      accr[s]  = wmma_bf16(ah, br1, accr[s]);
      accz[s]  = wmma_bf16(ax, bz0, accz[s]);
      accz[s]  = wmma_bf16(ah, bz1, accz[s]);
      accnx[s] = wmma_bf16(ax, bn0, accnx[s]);
      acchn[s] = wmma_bf16(ah, bn1, acchn[s]);
    }
    __syncthreads();                            // done reading buf before restage
    buf ^= 1;
  }

#pragma unroll
  for (int s = 0; s < 4; ++s) {
    const int n = n0 + 16 * s + l;
    const float br = bxr[n] + bhr[n];
    const float bz = bxz[n] + bhz[n];
    const float bn_x = bxn[n], bn_h = bhn[n];
#pragma unroll
    for (int j = 0; j < 8; ++j) {
      const int m = m0 + j + 8 * h;
      const float rr = sigmoidf_fast(accr[s][j] + br);
      const float zz = sigmoidf_fast(accz[s][j] + bz);
      const float nv = tanhf(accnx[s][j] + bn_x + rr * (acchn[s][j] + bn_h));
      const float hin = hidden[(size_t)m * H_DIM + n];  // fp32 for the z*h term
      const float hv = (1.0f - zz) * nv + zz * hin;
      h_out[(size_t)m * H_DIM + n] = hv;                 // fp32 output (tuple elem 1)
      h_bf[(size_t)m * H_DIM + n] = f32_to_bf16_rne(hv); // feed fc2
    }
  }
}

// ---------------------------------------------------------------------------
// Stage 3: q = 10 * tanh(h @ W2.T + b2). Wave tile 16 x 128 (all of A_DIM).
// W2 tile (128 x 32 per chunk) staged by the TDM, double buffered.
__global__ __launch_bounds__(256) void fc2_tanh_kernel(
    const u16* __restrict__ h_bf, const u16* __restrict__ W2bf,
    const float* __restrict__ b2, float* __restrict__ q) {
  __shared__ __align__(128) u16 sW[2][128][LDSPITCH];   // 20,480 B

  const int wave = threadIdx.x >> 5;
  const int lane = threadIdx.x & 31;
  const int l = lane & 15, h = lane >> 4;
  const int m0 = blockIdx.x * 128 + wave * 16;

  const u16* pa = h_bf + (size_t)(m0 + l) * H_DIM + 8 * h;
  const int kb = h << 3;

  if (wave == 0)
    tdm_load_tile_2d(W2bf, (unsigned)(size_t)&sW[0][0][0], 128, H_DIM);

  v8f acc[8] = {};
  int buf = 0;
  for (int kc = 0; kc < H_DIM; kc += 32) {
    const bool has_next = (kc + 32) < H_DIM;
    if (wave == 0) {
      if (has_next) {
        tdm_load_tile_2d(W2bf + kc + 32, (unsigned)(size_t)&sW[buf ^ 1][0][0],
                         128, H_DIM);
        __builtin_amdgcn_s_wait_tensorcnt(1);
      } else {
        __builtin_amdgcn_s_wait_tensorcnt(0);
      }
    }
    __syncthreads();

    v16bf a = frag_a_from(pa + kc);
    v16bf bf[8];
#pragma unroll
    for (int s = 0; s < 8; ++s) bf[s] = frag_b_from(&sW[buf][16 * s + l][kb]);
#pragma unroll
    for (int s = 0; s < 8; ++s) acc[s] = wmma_bf16(a, bf[s], acc[s]);

    __syncthreads();
    buf ^= 1;
  }

#pragma unroll
  for (int s = 0; s < 8; ++s) {
    const int n = 16 * s + l;
    const float bias = b2[n];
#pragma unroll
    for (int j = 0; j < 8; ++j) {
      const int m = m0 + j + 8 * h;
      q[(size_t)m * A_DIM + n] = RMAXV * tanhf(acc[s][j] + bias);
    }
  }
}

// ---------------------------------------------------------------------------
extern "C" void kernel_launch(void* const* d_in, const int* in_sizes, int n_in,
                              void* d_out, int out_size, void* d_ws, size_t ws_size,
                              hipStream_t stream) {
  const float* inputs = (const float*)d_in[0];
  const float* hidden = (const float*)d_in[1];
  const float* W1  = (const float*)d_in[2];  const float* b1  = (const float*)d_in[3];
  const float* Wxr = (const float*)d_in[4];  const float* bxr = (const float*)d_in[5];
  const float* Whr = (const float*)d_in[6];  const float* bhr = (const float*)d_in[7];
  const float* Wxz = (const float*)d_in[8];  const float* bxz = (const float*)d_in[9];
  const float* Whz = (const float*)d_in[10]; const float* bhz = (const float*)d_in[11];
  const float* Wxn = (const float*)d_in[12]; const float* bxn = (const float*)d_in[13];
  const float* Whn = (const float*)d_in[14]; const float* bhn = (const float*)d_in[15];
  const float* W2  = (const float*)d_in[16]; const float* b2  = (const float*)d_in[17];

  // Output tuple (q, h) flattened: q first, then h.
  float* q_out = (float*)d_out;
  float* h_out = (float*)d_out + (size_t)B_ROWS * A_DIM;

  // Workspace layout (bf16 elements). ~238 MB total.
  u16* ws = (u16*)d_ws;
  size_t off = 0;
  u16* in_bf  = ws + off; off += (size_t)B_ROWS * IN_DIM;
  u16* hid_bf = ws + off; off += (size_t)B_ROWS * H_DIM;
  u16* x_bf   = ws + off; off += (size_t)B_ROWS * H_DIM;
  u16* h_bf   = ws + off; off += (size_t)B_ROWS * H_DIM;
  u16* W1bf   = ws + off; off += (size_t)H_DIM * IN_DIM;
  u16* Wxrbf  = ws + off; off += (size_t)H_DIM * H_DIM;
  u16* Whrbf  = ws + off; off += (size_t)H_DIM * H_DIM;
  u16* Wxzbf  = ws + off; off += (size_t)H_DIM * H_DIM;
  u16* Whzbf  = ws + off; off += (size_t)H_DIM * H_DIM;
  u16* Wxnbf  = ws + off; off += (size_t)H_DIM * H_DIM;
  u16* Whnbf  = ws + off; off += (size_t)H_DIM * H_DIM;
  u16* W2bf   = ws + off; off += (size_t)A_DIM * H_DIM;

  auto cvt = [&](const float* s, u16* d, size_t n) {
    int n4 = (int)(n / 4);
    cvt_f32_to_bf16_kernel<<<dim3((n4 + 255) / 256), dim3(256), 0, stream>>>(s, d, n4);
  };
  cvt(inputs, in_bf,  (size_t)B_ROWS * IN_DIM);
  cvt(hidden, hid_bf, (size_t)B_ROWS * H_DIM);
  cvt(W1,  W1bf,  (size_t)H_DIM * IN_DIM);
  cvt(Wxr, Wxrbf, (size_t)H_DIM * H_DIM);
  cvt(Whr, Whrbf, (size_t)H_DIM * H_DIM);
  cvt(Wxz, Wxzbf, (size_t)H_DIM * H_DIM);
  cvt(Whz, Whzbf, (size_t)H_DIM * H_DIM);
  cvt(Wxn, Wxnbf, (size_t)H_DIM * H_DIM);
  cvt(Whn, Whnbf, (size_t)H_DIM * H_DIM);
  cvt(W2,  W2bf,  (size_t)A_DIM * H_DIM);

  dim3 blk(256);
  fc1_relu_kernel<<<dim3(B_ROWS / 128, H_DIM / 128), blk, 0, stream>>>(
      in_bf, W1bf, b1, x_bf);
  gru_fused_kernel<<<dim3(B_ROWS / 128, H_DIM / 64), blk, 0, stream>>>(
      x_bf, hid_bf, hidden, Wxrbf, Whrbf, Wxzbf, Whzbf, Wxnbf, Whnbf,
      bxr, bhr, bxz, bhz, bxn, bhn, h_out, h_bf);
  fc2_tanh_kernel<<<dim3(B_ROWS / 128), blk, 0, stream>>>(h_bf, W2bf, b2, q_out);
}
